// Model_30975304138830
// MI455X (gfx1250) — matrix operand order repaired and checked by reading, not compiled, not run on hardware
//
#include <hip/hip_runtime.h>
#include <math.h>

// ---------------- problem constants ----------------
#define LL 2
#define BB 2
#define SS 512
#define HH 1280
#define GG (4*HH)      // 5120
#define VV 32000
#define MM (BB*SS)     // 1024

typedef float v8f  __attribute__((ext_vector_type(8)));
typedef __bf16 v8bf  __attribute__((ext_vector_type(8)));
typedef __bf16 v16bf __attribute__((ext_vector_type(16)));

// ======== fp32-accurate GEMM via bf16-split WMMA: C = A*B + bias ==========
// A,B,C fp32 in memory. Per 32-k chunk, A and B are split on the fly into
// (hi, lo) bf16 tiles in LDS; acc += Ahi*Bhi + Ahi*Blo + Alo*Bhi  (3x
// v_wmma_f32_16x16x32_bf16, dropped lo*lo term ~2^-16 relative).
#define BM 128
#define BN 128
#define BKC 32          // k per chunk (one bf16 WMMA K)
#define LDK 40          // padded row stride in bf16 elems (16B-aligned frags, conflict-free)

__device__ __forceinline__ void splitbf(float x, unsigned short& h, unsigned short& l) {
  __bf16 hb = (__bf16)x;
  __bf16 lb = (__bf16)(x - (float)hb);
  h = __builtin_bit_cast(unsigned short, hb);
  l = __builtin_bit_cast(unsigned short, lb);
}

__device__ __forceinline__ v16bf ldfrag(const unsigned short* p0, const unsigned short* p1) {
  v8bf a = *(const v8bf*)p0;   // 16B aligned
  v8bf b = *(const v8bf*)p1;
  return __builtin_shufflevector(a, b, 0,1,2,3,4,5,6,7,8,9,10,11,12,13,14,15);
}

__global__ __launch_bounds__(256) void gemm_bias_f32(
    const float* __restrict__ A, const float* __restrict__ Bw,
    const float* __restrict__ bias, float* __restrict__ C,
    int N, int K)
{
  __shared__ unsigned short aHiS[BM * LDK];   // [m][k], k contiguous
  __shared__ unsigned short aLoS[BM * LDK];
  __shared__ unsigned short bHiS[BN * LDK];   // [n][k], k contiguous
  __shared__ unsigned short bLoS[BN * LDK];

  const int tid  = threadIdx.x;
  const int bn   = blockIdx.x;
  const int bm   = blockIdx.y;
  const int wave = tid >> 5;
  const int lane = tid & 31;
  const int wm   = wave & 3;       // 0..3 (M)
  const int wn   = wave >> 2;      // 0..1 (N)
  const int lm   = lane & 15;
  const int khi  = lane >> 4;      // lane half selects K sub-ranges

  v8f acc[2][4];
  const v8f vz = {0.f,0.f,0.f,0.f,0.f,0.f,0.f,0.f};
#pragma unroll
  for (int mi = 0; mi < 2; ++mi)
#pragma unroll
    for (int nj = 0; nj < 4; ++nj) acc[mi][nj] = vz;

  // global->LDS mapping
  const int ar  = tid & 127;        // A row in tile
  const int akg = tid >> 7;         // A k-half (16 k's)
  const int bkr = tid & 31;         // B k row in chunk
  const int bcg = (tid >> 5) * 16;  // B col group (16 cols; == wave*16)

  const float* arow = A + (size_t)(bm * BM + ar) * K;
  unsigned int* aHW = (unsigned int*)aHiS;
  unsigned int* aLW = (unsigned int*)aLoS;

  for (int k0 = 0; k0 < K; k0 += BKC) {
    float4 av[4], bv[4];
#pragma unroll
    for (int j = 0; j < 4; ++j)
      av[j] = *(const float4*)(arow + k0 + akg * 16 + 4 * j);
    const float* bp = Bw + (size_t)(k0 + bkr) * N + bn * BN + bcg;
#pragma unroll
    for (int j = 0; j < 4; ++j)
      bv[j] = *(const float4*)(bp + 4 * j);
    if (k0 + BKC < K)
      __builtin_prefetch(Bw + (size_t)(k0 + BKC + bkr) * N + bn * BN + bcg, 0, 1);

    __syncthreads();
    {   // A: split + packed 32-bit stores, row ar, k = akg*16 .. +16
      const float ax[16] = {av[0].x,av[0].y,av[0].z,av[0].w, av[1].x,av[1].y,av[1].z,av[1].w,
                            av[2].x,av[2].y,av[2].z,av[2].w, av[3].x,av[3].y,av[3].z,av[3].w};
      const int wbase = ar * (LDK / 2) + akg * 8;
#pragma unroll
      for (int j = 0; j < 8; ++j) {
        unsigned short h0,l0,h1,l1;
        splitbf(ax[2*j],   h0, l0);
        splitbf(ax[2*j+1], h1, l1);
        aHW[wbase + j] = (unsigned int)h0 | ((unsigned int)h1 << 16);
        aLW[wbase + j] = (unsigned int)l0 | ((unsigned int)l1 << 16);
      }
    }
    {   // B: split + scatter (transpose k-major -> n-major)
      const float bx[16] = {bv[0].x,bv[0].y,bv[0].z,bv[0].w, bv[1].x,bv[1].y,bv[1].z,bv[1].w,
                            bv[2].x,bv[2].y,bv[2].z,bv[2].w, bv[3].x,bv[3].y,bv[3].z,bv[3].w};
#pragma unroll
      for (int i = 0; i < 16; ++i) {
        unsigned short h,l;
        splitbf(bx[i], h, l);
        bHiS[(bcg + i) * LDK + bkr] = h;
        bLoS[(bcg + i) * LDK + bkr] = l;
      }
    }
    __syncthreads();

    // 3-term accumulation: HH, H*L, L*H (fragments reloaded per term -> low reg pressure)
#pragma unroll
    for (int term = 0; term < 3; ++term) {
      const unsigned short* aSrc = (term == 2) ? aLoS : aHiS;
      const unsigned short* bSrc = (term == 1) ? bLoS : bHiS;
      v16bf af[2], bf[4];
#pragma unroll
      for (int mi = 0; mi < 2; ++mi) {
        const int m = wm * 32 + mi * 16 + lm;
        // A 16-bit layout: lanes 0-15 K=[0..7]+[16..23], lanes 16-31 K=[8..15]+[24..31]
        af[mi] = ldfrag(aSrc + m * LDK + khi * 8, aSrc + m * LDK + 16 + khi * 8);
      }
#pragma unroll
      for (int nj = 0; nj < 4; ++nj) {
        const int n = wn * 64 + nj * 16 + lm;
        // B 16-bit layout: lanes 0-15 K=[0..7]+[16..23], lanes 16-31 K=[8..15]+[24..31]
        bf[nj] = ldfrag(bSrc + n * LDK + khi * 8, bSrc + n * LDK + 16 + khi * 8);
      }
#pragma unroll
      for (int mi = 0; mi < 2; ++mi)
#pragma unroll
        for (int nj = 0; nj < 4; ++nj)
          acc[mi][nj] = __builtin_amdgcn_wmma_f32_16x16x32_bf16(
              false, af[mi], false, bf[nj], (short)0, acc[mi][nj], false, false);
    }
  }

  // epilogue: C[m][n] = acc + bias[n]  (nontemporal: outputs streamed once)
#pragma unroll
  for (int nj = 0; nj < 4; ++nj) {
    const int n = bn * BN + wn * 64 + nj * 16 + lm;
    const float bv = bias[n];
#pragma unroll
    for (int mi = 0; mi < 2; ++mi) {
      const int m0 = bm * BM + wm * 32 + mi * 16 + khi * 8;
#pragma unroll
      for (int i = 0; i < 8; ++i)
        __builtin_nontemporal_store(acc[mi][nj][i] + bv, C + (size_t)(m0 + i) * N + n);
    }
  }
}

// ================= utility =================
__global__ void zero_f32(float* p, int n) {
  int i = blockIdx.x * blockDim.x + threadIdx.x;
  if (i < n) p[i] = 0.f;
}

__device__ __forceinline__ float sigf(float x) { return 1.f / (1.f + __expf(-x)); }

// sense-reversing grid barrier (all 40 resident WGs of the persistent scan)
__device__ __forceinline__ void grid_barrier(unsigned* cnt, unsigned* gen, unsigned nb) {
  __syncthreads();
  if (threadIdx.x == 0) {
    __threadfence();
    unsigned g = __hip_atomic_load(gen, __ATOMIC_ACQUIRE, __HIP_MEMORY_SCOPE_AGENT);
    unsigned prev = __hip_atomic_fetch_add(cnt, 1u, __ATOMIC_ACQ_REL, __HIP_MEMORY_SCOPE_AGENT);
    if (prev == nb - 1u) {
      __hip_atomic_store(cnt, 0u, __ATOMIC_RELAXED, __HIP_MEMORY_SCOPE_AGENT);
      __hip_atomic_fetch_add(gen, 1u, __ATOMIC_RELEASE, __HIP_MEMORY_SCOPE_AGENT);
    } else {
      while (__hip_atomic_load(gen, __ATOMIC_ACQUIRE, __HIP_MEMORY_SCOPE_AGENT) == g)
        __builtin_amdgcn_s_sleep(1);
    }
    __threadfence();
  }
  __syncthreads();
}

// ================= persistent LSTM scan: both directions of one layer =================
// 40 WGs x 256 thr = 10240 threads. Thread owns (dir, cell cc, k-quarter ks).
// Phase1: g = xp + h@Ws (4 gate dots, both batches, K split over 4 lanes) -> cell update -> m
// Phase3: h_new = m@Wp  -> write h and residual output.
#define RNB 40
#define RTPB 256
#define KS 4
#define KL (HH / KS)   // 320

__global__ __launch_bounds__(RTPB) void lstm_scan(
    const float* __restrict__ xpF, const float* __restrict__ xpB,
    const float* __restrict__ WsF, const float* __restrict__ WsB,
    const float* __restrict__ WpF, const float* __restrict__ WpB,
    const float* __restrict__ resF, const float* __restrict__ resB, int addRes,
    float* __restrict__ outF, float* __restrict__ outB,
    float* __restrict__ hbuf, float* __restrict__ cbuf, float* __restrict__ mbuf,
    unsigned* __restrict__ bar)
{
  __shared__ float vec[2 * 2 * HH];   // staged h (then m): [dir][b][HH], 20 KB

  const int tid = threadIdx.x;
  const int gt  = blockIdx.x * RTPB + tid;          // 0..10239
  const int ks  = gt & (KS - 1);
  const int idx = gt >> 2;                          // 0..2559
  const int dir = idx / HH;                         // 0 fwd, 1 bwd
  const int cc  = idx - dir * HH;                   // cell / output index
  const int k0  = ks * KL;

  const float* Ws  = dir ? WsB : WsF;
  const float* Wp  = dir ? WpB : WpF;
  const float* xp  = dir ? xpB : xpF;
  const float* res = dir ? resB : resF;
  float*       out = dir ? outB : outF;

  unsigned* cnt = bar;
  unsigned* gen = bar + 1;

  for (int t = 0; t < SS; ++t) {
    const int tt = dir ? (SS - 1 - t) : t;

    // stage h into LDS
    for (int i = tid; i < 2 * 2 * HH; i += RTPB) vec[i] = hbuf[i];
    __syncthreads();

    // ---- phase 1: gate pre-activations (partial K) ----
    float s0=0.f,s1=0.f,s2=0.f,s3=0.f,s4=0.f,s5=0.f,s6=0.f,s7=0.f;
    {
      const float* wcol = Ws + cc;
      const float* h0 = &vec[(dir * 2 + 0) * HH];
      const float* h1 = &vec[(dir * 2 + 1) * HH];
      for (int k = k0; k < k0 + KL; ++k) {
        const float w0 = wcol[(size_t)k * GG];
        const float w1 = wcol[(size_t)k * GG + HH];
        const float w2 = wcol[(size_t)k * GG + 2 * HH];
        const float w3 = wcol[(size_t)k * GG + 3 * HH];
        const float ha = h0[k], hb = h1[k];
        s0 += ha * w0; s1 += ha * w1; s2 += ha * w2; s3 += ha * w3;
        s4 += hb * w0; s5 += hb * w1; s6 += hb * w2; s7 += hb * w3;
      }
    }
    // reduce the 4-lane K split
    s0 += __shfl_xor(s0,1); s0 += __shfl_xor(s0,2);
    s1 += __shfl_xor(s1,1); s1 += __shfl_xor(s1,2);
    s2 += __shfl_xor(s2,1); s2 += __shfl_xor(s2,2);
    s3 += __shfl_xor(s3,1); s3 += __shfl_xor(s3,2);
    s4 += __shfl_xor(s4,1); s4 += __shfl_xor(s4,2);
    s5 += __shfl_xor(s5,1); s5 += __shfl_xor(s5,2);
    s6 += __shfl_xor(s6,1); s6 += __shfl_xor(s6,2);
    s7 += __shfl_xor(s7,1); s7 += __shfl_xor(s7,2);

    if (ks == 0) {
      const float sums[2][4] = {{s0,s1,s2,s3},{s4,s5,s6,s7}};
      for (int b = 0; b < 2; ++b) {
        const float* xr = xp + (size_t)(b * SS + tt) * GG + cc;
        const float gi = xr[0]       + sums[b][0];
        const float gf = xr[HH]      + sums[b][1];
        const float gz = xr[2 * HH]  + sums[b][2];
        const float go = xr[3 * HH]  + sums[b][3];
        const int ci = (dir * 2 + b) * HH + cc;
        const float cn = sigf(gi) * tanhf(gz) + sigf(gf) * cbuf[ci];
        cbuf[ci] = cn;
        mbuf[ci] = sigf(go) * tanhf(cn);
      }
    }
    grid_barrier(cnt, gen, RNB);

    // stage m into LDS
    for (int i = tid; i < 2 * 2 * HH; i += RTPB) vec[i] = mbuf[i];
    __syncthreads();

    // ---- phase 3: projection h_new = m @ Wp (partial K) ----
    float p0 = 0.f, p1 = 0.f;
    {
      const float* wcol = Wp + cc;
      const float* m0 = &vec[(dir * 2 + 0) * HH];
      const float* m1 = &vec[(dir * 2 + 1) * HH];
      for (int k = k0; k < k0 + KL; ++k) {
        const float w = wcol[(size_t)k * HH];
        p0 += m0[k] * w;
        p1 += m1[k] * w;
      }
    }
    p0 += __shfl_xor(p0,1); p0 += __shfl_xor(p0,2);
    p1 += __shfl_xor(p1,1); p1 += __shfl_xor(p1,2);

    if (ks == 0) {
      hbuf[(dir * 2 + 0) * HH + cc] = p0;
      hbuf[(dir * 2 + 1) * HH + cc] = p1;
      float o0 = p0, o1 = p1;
      if (addRes) {
        o0 += res[(size_t)(0 * SS + tt) * HH + cc];
        o1 += res[(size_t)(1 * SS + tt) * HH + cc];
      }
      out[(size_t)(0 * SS + tt) * HH + cc] = o0;
      out[(size_t)(1 * SS + tt) * HH + cc] = o1;
    }
    grid_barrier(cnt, gen, RNB);
  }
}

// ================= host orchestration =================
extern "C" void kernel_launch(void* const* d_in, const int* in_sizes, int n_in,
                              void* d_out, int out_size, void* d_ws, size_t ws_size,
                              hipStream_t stream) {
  (void)in_sizes; (void)n_in; (void)out_size; (void)ws_size;
  const float* x     = (const float*)d_in[0];
  const float* Wi_f  = (const float*)d_in[1];
  const float* Ws_f  = (const float*)d_in[2];
  const float* b_f   = (const float*)d_in[3];
  const float* Wp_f  = (const float*)d_in[4];
  const float* Wi_b  = (const float*)d_in[5];
  const float* Ws_b  = (const float*)d_in[6];
  const float* b_b   = (const float*)d_in[7];
  const float* Wp_b  = (const float*)d_in[8];
  const float* W_lin = (const float*)d_in[9];
  const float* b_lin = (const float*)d_in[10];
  float* outp = (float*)d_out;

  // workspace layout (floats)
  float* ws = (float*)d_ws;
  const size_t XPSZ  = (size_t)MM * GG;   // 5,242,880
  const size_t OUTSZ = (size_t)MM * HH;   // 1,310,720
  float* xpF  = ws;
  float* xpB  = xpF + XPSZ;
  float* fo0  = xpB + XPSZ;
  float* fo1  = fo0 + OUTSZ;
  float* bo0  = fo1 + OUTSZ;
  float* bo1  = bo0 + OUTSZ;
  float* hbuf = bo1 + OUTSZ;
  float* cbuf = hbuf + 2 * 2 * HH;
  float* mbuf = cbuf + 2 * 2 * HH;
  unsigned* bar = (unsigned*)(mbuf + 2 * 2 * HH);
  const int stateWords = 3 * (2 * 2 * HH) + 4;   // h, c, m, barrier

  float* foL[2] = {fo0, fo1};
  float* boL[2] = {bo0, bo1};

  for (int l = 0; l < LL; ++l) {
    const float* Xf = (l == 0) ? x : foL[l - 1];
    const float* Xb = (l == 0) ? x : boL[l - 1];

    // input projections xp = X @ Wi + b  (both directions)
    gemm_bias_f32<<<dim3(GG / BN, MM / BM), 256, 0, stream>>>(
        Xf, Wi_f + (size_t)l * HH * GG, b_f + (size_t)l * GG, xpF, GG, HH);
    gemm_bias_f32<<<dim3(GG / BN, MM / BM), 256, 0, stream>>>(
        Xb, Wi_b + (size_t)l * HH * GG, b_b + (size_t)l * GG, xpB, GG, HH);

    // reset h/c/m + barrier state (d_ws is poisoned; must be deterministic each call)
    zero_f32<<<(stateWords + 255) / 256, 256, 0, stream>>>(hbuf, stateWords);

    // persistent cooperative recurrence, fwd+bwd together
    lstm_scan<<<RNB, RTPB, 0, stream>>>(
        xpF, xpB,
        Ws_f + (size_t)l * HH * GG, Ws_b + (size_t)l * HH * GG,
        Wp_f + (size_t)l * HH * HH, Wp_b + (size_t)l * HH * HH,
        Xf, Xb, (l > 0) ? 1 : 0,
        foL[l], boL[l], hbuf, cbuf, mbuf, bar);
  }

  // vocab projection: logits = enc @ W_lin + b_lin
  // layout: fwd (l,b,s,v) at l*MM*VV ; bwd at (2+l)*MM*VV
  for (int l = 0; l < LL; ++l)
    gemm_bias_f32<<<dim3(VV / BN, MM / BM), 256, 0, stream>>>(
        foL[l], W_lin, b_lin, outp + (size_t)l * MM * VV, VV, HH);
  for (int l = 0; l < LL; ++l)
    gemm_bias_f32<<<dim3(VV / BN, MM / BM), 256, 0, stream>>>(
        boL[l], W_lin, b_lin, outp + (size_t)(LL + l) * MM * VV, VV, HH);
}
